// AFMADEBlock_63393717289566
// MI455X (gfx1250) — matrix operand order repaired and checked by reading, not compiled, not run on hardware
//
#include <hip/hip_runtime.h>

typedef __bf16 bf16_t;
typedef __attribute__((ext_vector_type(16))) __bf16 v16bf;
typedef __attribute__((ext_vector_type(8)))  __bf16 v8bf;
typedef __attribute__((ext_vector_type(8)))  float  v8f;
typedef __attribute__((ext_vector_type(4)))  int    v4i_t;

#define DD   32      // input size
#define HH   256     // hidden size
#define BB   4096    // batch
#define ROWS 128     // batch rows per workgroup (8 waves x 16 rows)
#define HSTR 264     // padded LDS stride (bank-conflict-free b128 lanes)
#define YSTR 36      // padded y-row stride (floats)
#define EPS  1e-12f

// workspace layout (bf16 elements), column-major [fo][fi] per matrix:
#define W0_OFF 0
#define W1_OFF 8192
#define W2_OFF 73728
#define NET_SZ 81920

// Async global->LDS staging (gfx1250 ASYNCcnt path) when the toolchain has it.
#if defined(__has_builtin)
#if __has_builtin(__builtin_amdgcn_global_load_async_to_lds_b128) && \
    __has_builtin(__builtin_amdgcn_s_wait_asynccnt)
#define USE_ASYNC_LDS 1
#endif
#endif
#ifndef USE_ASYNC_LDS
#define USE_ASYNC_LDS 0
#endif

#define GLOBAL_AS __attribute__((address_space(1)))
#define LDS_AS    __attribute__((address_space(3)))

// ---------------------------------------------------------------------------
// Prep: apply MADE masks, convert to bf16, store column-major so WMMA B
// fragments are contiguous 32B runs along K.
// ---------------------------------------------------------------------------
__global__ void afmade_prep(const float* __restrict__ muW0,
                            const float* __restrict__ muW1,
                            const float* __restrict__ muW2,
                            const float* __restrict__ lvW0,
                            const float* __restrict__ lvW1,
                            const float* __restrict__ lvW2,
                            bf16_t* __restrict__ ws) {
    int t = blockIdx.x * blockDim.x + threadIdx.x;
    if (t >= NET_SZ) return;
    float m, vmu, vlv;
    if (t < W1_OFF) {                     // layer0: fi=32, fo=256
        int o = t >> 5, i = t & 31;
        int degh = (o % (DD - 1)) + 1, degi = i + 1;
        m = (degh >= degi) ? 1.f : 0.f;
        vmu = muW0[i * HH + o]; vlv = lvW0[i * HH + o];
    } else if (t < W2_OFF) {              // layer1: fi=256, fo=256
        int u = t - W1_OFF;
        int o = u >> 8, i = u & 255;
        int dgo = (o % (DD - 1)) + 1, dgi = (i % (DD - 1)) + 1;
        m = (dgo >= dgi) ? 1.f : 0.f;
        vmu = muW1[i * HH + o]; vlv = lvW1[i * HH + o];
    } else {                              // layer2: fi=256, fo=32 (strict)
        int u = t - W2_OFF;
        int o = u >> 8, i = u & 255;
        int dgo = o + 1, dgi = (i % (DD - 1)) + 1;
        m = (dgo > dgi) ? 1.f : 0.f;
        vmu = muW2[i * DD + o]; vlv = lvW2[i * DD + o];
    }
    ws[t]          = (bf16_t)(vmu * m);
    ws[NET_SZ + t] = (bf16_t)(vlv * m);
}

// ---------------------------------------------------------------------------
// Fragment helpers (CDNA5 wave32 WMMA layouts, CDNA5 ISA)
// ---------------------------------------------------------------------------
__device__ __forceinline__ v16bf ld_frag2(const bf16_t* p0, const bf16_t* p1) {
    union { v16bf v; v8bf h[2]; } u;
    u.h[0] = *(const v8bf*)p0;   // 16B aligned
    u.h[1] = *(const v8bf*)p1;
    return u.v;
}

// A 16x32 bf16 from LDS row-major (stride HSTR)
__device__ __forceinline__ v16bf load_a_lds(const bf16_t* hrow, int kk, int koff) {
    return ld_frag2(hrow + kk + koff, hrow + kk + koff + 8);
}

// B 32x16 from padded LDS W1 (column stride HSTR): 16 contiguous halves.
__device__ __forceinline__ v16bf load_b_lds(const bf16_t* w1l, int ncol, int kb) {
    const bf16_t* p = w1l + ncol * HSTR + kb;
    return ld_frag2(p, p + 8);
}

// B 32x16 from global column-major weights (32B aligned contiguous run).
__device__ __forceinline__ v16bf load_b_glb(const bf16_t* __restrict__ wcm,
                                            int K, int ncol, int kb) {
    return *(const v16bf*)(wcm + (size_t)ncol * K + kb);
}

#define WMMA_BF16(A, B, C) \
    __builtin_amdgcn_wmma_f32_16x16x32_bf16(false, (A), false, (B), (short)0, (C), false, false)

// ---------------------------------------------------------------------------
// One masked MLP (D->H->H->D) for this wave's 16 rows. W1 comes from LDS.
// N-tiles processed two-at-a-time: the second tile's WMMA fills the first
// tile's WMMA->VALU hazard slots; unroll(disable) bounds load hoisting so
// nothing spills to scratch.
// ---------------------------------------------------------------------------
__device__ __forceinline__ void run_net(v16bf a0,
                                        const bf16_t* __restrict__ w0g,
                                        const bf16_t* __restrict__ w2g,
                                        const bf16_t* w1l,
                                        const float* __restrict__ b0,
                                        const float* __restrict__ b1,
                                        const float* __restrict__ b2,
                                        bf16_t* hA, bf16_t* hB,
                                        int rbase, int lane,
                                        v8f* o0, v8f* o1) {
    const int n     = lane & 15;
    const int rr    = (lane >> 4) * 8;
    const int koff  = (lane >> 4) * 16;
    const int kboff = (lane >> 4) * 16;

    // ---- layer 0: [16x32] @ [32x256] (B streamed from L2) ----
#pragma clang loop unroll(disable)
    for (int nt = 0; nt < 16; nt += 2) {
        v16bf bb0 = load_b_glb(w0g, DD, (nt + 0) * 16 + n, kboff);
        v16bf bb1 = load_b_glb(w0g, DD, (nt + 1) * 16 + n, kboff);
        v8f acc0 = {}, acc1 = {};
        acc0 = WMMA_BF16(a0, bb0, acc0);
        acc1 = WMMA_BF16(a0, bb1, acc1);
        float bias0 = b0[(nt + 0) * 16 + n];
        float bias1 = b0[(nt + 1) * 16 + n];
#pragma unroll
        for (int r = 0; r < 8; ++r) {
            float v0 = acc0[r] + bias0;  v0 = v0 > 0.f ? v0 : 0.f;
            float v1 = acc1[r] + bias1;  v1 = v1 > 0.f ? v1 : 0.f;
            hA[(rbase + rr + r) * HSTR + (nt + 0) * 16 + n] = (bf16_t)v0;
            hA[(rbase + rr + r) * HSTR + (nt + 1) * 16 + n] = (bf16_t)v1;
        }
    }

    // ---- layer 1: [16x256] @ [256x256] (B from LDS stage) ----
    {
        const bf16_t* arow = hA + (rbase + n) * HSTR;
        v16bf ac[8];
#pragma unroll
        for (int k8 = 0; k8 < 8; ++k8) ac[k8] = load_a_lds(arow, k8 * 32, koff);
#pragma clang loop unroll(disable)
        for (int nt = 0; nt < 16; nt += 2) {
            v8f acc0 = {}, acc1 = {};
#pragma unroll
            for (int k8 = 0; k8 < 8; ++k8) {
                v16bf bb0 = load_b_lds(w1l, (nt + 0) * 16 + n, k8 * 32 + kboff);
                v16bf bb1 = load_b_lds(w1l, (nt + 1) * 16 + n, k8 * 32 + kboff);
                acc0 = WMMA_BF16(ac[k8], bb0, acc0);
                acc1 = WMMA_BF16(ac[k8], bb1, acc1);
            }
            float bias0 = b1[(nt + 0) * 16 + n];
            float bias1 = b1[(nt + 1) * 16 + n];
#pragma unroll
            for (int r = 0; r < 8; ++r) {
                float v0 = acc0[r] + bias0;  v0 = v0 > 0.f ? v0 : 0.f;
                float v1 = acc1[r] + bias1;  v1 = v1 > 0.f ? v1 : 0.f;
                hB[(rbase + rr + r) * HSTR + (nt + 0) * 16 + n] = (bf16_t)v0;
                hB[(rbase + rr + r) * HSTR + (nt + 1) * 16 + n] = (bf16_t)v1;
            }
        }
    }

    // ---- layer 2: [16x256] @ [256x32], no relu (B streamed from L2) ----
    {
        const bf16_t* arow = hB + (rbase + n) * HSTR;
        v16bf ac[8];
#pragma unroll
        for (int k8 = 0; k8 < 8; ++k8) ac[k8] = load_a_lds(arow, k8 * 32, koff);
        v8f acc0 = {}, acc1 = {};
#pragma unroll
        for (int k8 = 0; k8 < 8; ++k8) {
            v16bf bb0 = load_b_glb(w2g, HH, 0 * 16 + n, k8 * 32 + kboff);
            v16bf bb1 = load_b_glb(w2g, HH, 1 * 16 + n, k8 * 32 + kboff);
            acc0 = WMMA_BF16(ac[k8], bb0, acc0);
            acc1 = WMMA_BF16(ac[k8], bb1, acc1);
        }
        float bias0 = b2[0 * 16 + n];
        float bias1 = b2[1 * 16 + n];
#pragma unroll
        for (int r = 0; r < 8; ++r) { acc0[r] += bias0; acc1[r] += bias1; }
        *o0 = acc0;
        *o1 = acc1;
    }
}

// ---------------------------------------------------------------------------
// Cooperative stage of one net's W1 (column-major 256x256 bf16) into padded
// LDS; thread t copies column t (32 x 16B), shared by all 8 waves.
// Uses the gfx1250 async global->LDS engine when available (no VGPR round
// trip, tracked by ASYNCcnt); falls back to load+store otherwise.
// ---------------------------------------------------------------------------
#if USE_ASYNC_LDS
__device__ __forceinline__ void stage_w1(const bf16_t* __restrict__ src,
                                         bf16_t* dst, int tid) {
    GLOBAL_AS v4i_t* s = (GLOBAL_AS v4i_t*)(src + (size_t)tid * HH);
    LDS_AS    v4i_t* d = (LDS_AS    v4i_t*)(dst + tid * HSTR);
#pragma unroll
    for (int j = 0; j < 32; ++j) {
        __builtin_amdgcn_global_load_async_to_lds_b128(
            s + j, d + j, /*offset=*/0, /*cpol=*/0);
    }
}
__device__ __forceinline__ void stage_wait() {
    __builtin_amdgcn_s_wait_asynccnt(0);   // LDS writes landed for this wave
}
#else
__device__ __forceinline__ void stage_w1(const bf16_t* __restrict__ src,
                                         bf16_t* dst, int tid) {
    const bf16_t* s = src + (size_t)tid * HH;
    bf16_t*       d = dst + tid * HSTR;
#pragma unroll 8
    for (int j = 0; j < 32; ++j)
        *(uint4*)(d + j * 8) = *(const uint4*)(s + j * 8);
}
__device__ __forceinline__ void stage_wait() {}
#endif

// ---------------------------------------------------------------------------
__global__ void __launch_bounds__(256)
afmade_kernel(const float* __restrict__ x,
              const float* __restrict__ mu_b0, const float* __restrict__ mu_b1,
              const float* __restrict__ mu_b2,
              const float* __restrict__ lv_b0, const float* __restrict__ lv_b1,
              const float* __restrict__ lv_b2,
              const bf16_t* __restrict__ ws, float* __restrict__ out) {
    __shared__ __align__(16) bf16_t wl1[HH * HSTR];    // 132 KB staged W1
    __shared__ __align__(16) bf16_t hA[ROWS * HSTR];   // 66 KB
    __shared__ __align__(16) bf16_t hB[ROWS * HSTR];   // 66 KB
    __shared__ __align__(16) float  yld[ROWS * YSTR];  // 18 KB

    const int tid   = threadIdx.x;
    const int wave  = tid >> 5;
    const int lane  = tid & 31;
    const int rbase = wave * 16;
    const int gbase = blockIdx.x * ROWS + rbase;
    const int n     = lane & 15;
    const int rr    = (lane >> 4) * 8;
    const int koff  = (lane >> 4) * 16;

    const bf16_t* mu_w0 = ws + W0_OFF;
    const bf16_t* mu_w1 = ws + W1_OFF;
    const bf16_t* mu_w2 = ws + W2_OFF;
    const bf16_t* lv_w0 = ws + NET_SZ + W0_OFF;
    const bf16_t* lv_w1 = ws + NET_SZ + W1_OFF;
    const bf16_t* lv_w2 = ws + NET_SZ + W2_OFF;

    // y starts at zero
#pragma unroll
    for (int r = 0; r < 16; ++r) yld[(rbase + r) * YSTR + lane] = 0.f;

    v8f mu0, mu1, lv0, lv1;
    for (int stepi = 0; stepi < DD; ++stepi) {
        // layer-0 A fragment from f32 y (shared by both nets): K=32
        const float* yr = yld + (rbase + n) * YSTR;
        union { v16bf v; bf16_t e[16]; } a0u;
#pragma unroll
        for (int j = 0; j < 8; ++j) {
            a0u.e[j]     = (bf16_t)yr[koff + j];
            a0u.e[8 + j] = (bf16_t)yr[koff + 8 + j];
        }

        // ---- mu net ----
        stage_w1(mu_w1, wl1, tid);
        stage_wait();
        __syncthreads();
        __builtin_prefetch(lv_w1 + (size_t)tid * HH, 0, 1);  // warm lv W1
        run_net(a0u.v, mu_w0, mu_w2, wl1, mu_b0, mu_b1, mu_b2,
                hA, hB, rbase, lane, &mu0, &mu1);
        __syncthreads();          // all waves done reading staged mu W1

        // ---- lv net ----
        stage_w1(lv_w1, wl1, tid);
        stage_wait();
        __syncthreads();
        run_net(a0u.v, lv_w0, lv_w2, wl1, lv_b0, lv_b1, lv_b2,
                hA, hB, rbase, lane, &lv0, &lv1);

        // y_new = (x - mu) / (exp(0.5*lv) + eps), in C/D register layout
        const bool last = (stepi == DD - 1);
#pragma unroll
        for (int t2 = 0; t2 < 2; ++t2) {
            v8f mu = t2 ? mu1 : mu0;
            v8f lv = t2 ? lv1 : lv0;
            int col = t2 * 16 + n;
#pragma unroll
            for (int r = 0; r < 8; ++r) {
                int grow = gbase + rr + r;
                float ls = 0.5f * lv[r];
                float yn = (x[grow * DD + col] - mu[r]) / (__expf(ls) + EPS);
                yld[(rbase + rr + r) * YSTR + col] = yn;
                if (last) out[grow * DD + col] = yn;
            }
        }
        __syncthreads();          // staged lv W1 reads done before next stage
    }

    // logstd.sum(axis=1) of the final step; reduce across 16-lane half-wave.
    float s[8];
#pragma unroll
    for (int r = 0; r < 8; ++r) s[r] = 0.5f * (lv0[r] + lv1[r]);
#pragma unroll
    for (int m = 1; m < 16; m <<= 1) {
#pragma unroll
        for (int r = 0; r < 8; ++r) s[r] += __shfl_xor(s[r], m, 32);
    }
    if (n == 0) {
#pragma unroll
        for (int r = 0; r < 8; ++r) out[BB * DD + gbase + rr + r] = s[r];
    }
}

// ---------------------------------------------------------------------------
extern "C" void kernel_launch(void* const* d_in, const int* in_sizes, int n_in,
                              void* d_out, int out_size, void* d_ws, size_t ws_size,
                              hipStream_t stream) {
    const float* x     = (const float*)d_in[0];
    const float* muW0  = (const float*)d_in[1];
    const float* mu_b0 = (const float*)d_in[2];
    const float* muW1  = (const float*)d_in[3];
    const float* mu_b1 = (const float*)d_in[4];
    const float* muW2  = (const float*)d_in[5];
    const float* mu_b2 = (const float*)d_in[6];
    const float* lvW0  = (const float*)d_in[7];
    const float* lv_b0 = (const float*)d_in[8];
    const float* lvW1  = (const float*)d_in[9];
    const float* lv_b1 = (const float*)d_in[10];
    const float* lvW2  = (const float*)d_in[11];
    const float* lv_b2 = (const float*)d_in[12];
    bf16_t* ws  = (bf16_t*)d_ws;
    float*  out = (float*)d_out;

    afmade_prep<<<NET_SZ / 256, 256, 0, stream>>>(muW0, muW1, muW2,
                                                  lvW0, lvW1, lvW2, ws);
    afmade_kernel<<<BB / ROWS, 256, 0, stream>>>(x, mu_b0, mu_b1, mu_b2,
                                                 lv_b0, lv_b1, lv_b2, ws, out);
}
